// TopK_custom_5471788335850
// MI455X (gfx1250) — compile-verified, matrix operand order checked by reading, not run on hardware
//
#include <hip/hip_runtime.h>
#include <hip/hip_bf16.h>

typedef __attribute__((ext_vector_type(16))) _Float16 v16h;
typedef __attribute__((ext_vector_type(8)))  float    v8f;

#define SK_EPS   0.001f
#define SK_LOG2E 1.4426950408889634f

__device__ __forceinline__ float fast_exp2(float x) {
#if __has_builtin(__builtin_amdgcn_exp2f)
  return __builtin_amdgcn_exp2f(x);
#else
  return exp2f(x);
#endif
}
__device__ __forceinline__ float fast_log2(float x) {
#if __has_builtin(__builtin_amdgcn_logf)
  return __builtin_amdgcn_logf(x);
#else
  return log2f(x);
#endif
}
__device__ __forceinline__ float fast_rcp(float x) {
#if __has_builtin(__builtin_amdgcn_rcpf)
  return __builtin_amdgcn_rcpf(x);
#else
  return 1.0f / x;
#endif
}
__device__ __forceinline__ float neg_inf_f() { return -__builtin_inff(); }

// ---------------- Kernel 1: per-block max / min(excl -inf) / raw min ----------------
__global__ __launch_bounds__(256) void sk_minmax(const float* __restrict__ x,
                                                 long long total,
                                                 float* __restrict__ ws) {
  __shared__ float smax[256], smine[256], sraw[256];
  const int t = threadIdx.x;
  float mx = -__builtin_inff();
  float mne = __builtin_inff();   // min excluding -inf
  float mraw = __builtin_inff();  // plain min (detects presence of -inf)
  for (long long i = (long long)blockIdx.x * 256 + t; i < total;
       i += (long long)gridDim.x * 256) {
    float v = x[i];
    mx = fmaxf(mx, v);
    mraw = fminf(mraw, v);
    float ve = (v == neg_inf_f()) ? __builtin_inff() : v;
    mne = fminf(mne, ve);
  }
  smax[t] = mx; smine[t] = mne; sraw[t] = mraw;
  __syncthreads();
  for (int s = 128; s > 0; s >>= 1) {
    if (t < s) {
      smax[t]  = fmaxf(smax[t],  smax[t + s]);
      smine[t] = fminf(smine[t], smine[t + s]);
      sraw[t]  = fminf(sraw[t],  sraw[t + s]);
    }
    __syncthreads();
  }
  if (t == 0) {
    ws[16 + blockIdx.x]        = smax[0];
    ws[16 + 256 + blockIdx.x]  = smine[0];
    ws[16 + 512 + blockIdx.x]  = sraw[0];
  }
}

// ---------------- Kernel 2: fold partials -> filled, Q = log2e/(Cmax*eps) ----------------
__global__ __launch_bounds__(256) void sk_scalars(float* __restrict__ ws) {
  __shared__ float smax[256], smine[256], sraw[256];
  const int t = threadIdx.x;
  smax[t]  = ws[16 + t];
  smine[t] = ws[16 + 256 + t];
  sraw[t]  = ws[16 + 512 + t];
  __syncthreads();
  for (int s = 128; s > 0; s >>= 1) {
    if (t < s) {
      smax[t]  = fmaxf(smax[t],  smax[t + s]);
      smine[t] = fminf(smine[t], smine[t + s]);
      sraw[t]  = fminf(sraw[t],  sraw[t + s]);
    }
    __syncthreads();
  }
  if (t == 0) {
    float max_s = smax[0];
    float min_s = smine[0];
    float raw   = sraw[0];
    float filled = min_s - (max_s - min_s);
    bool has_neg = (raw == neg_inf_f());
    float eff_min = has_neg ? filled : min_s;  // filled <= min_s
    // Cmax = max over anchors {0,1} of squared distance at extreme scores
    float c0 = fmaxf(max_s * max_s, eff_min * eff_min);
    float m1 = max_s - 1.0f, e1 = eff_min - 1.0f;
    float c1 = fmaxf(m1 * m1, e1 * e1);
    float Cmax = fmaxf(c0, c1);
    float Q = SK_LOG2E / (Cmax * SK_EPS);  // d2_i = (2*s_i - 1) * Q  (log2 domain)
    ws[0] = filled;
    ws[1] = Q;
  }
}

// ---------------- Kernel 3: per-row LDS-resident Sinkhorn (scalar-c iteration) ----------------
// One 1024-thread workgroup (32 waves) per batch row. Row's d2 values live in LDS.
// Per iteration: S = sum_i sigmoid(c - d_i), accumulated per-wave via
// V_WMMA_F32_16X16X32_F16 with an all-ones B matrix (512 values per WMMA).
__global__ __launch_bounds__(1024) void sk_rows(const float* __restrict__ scores,
                                                const int* __restrict__ kptr,
                                                float* __restrict__ out,
                                                const float* __restrict__ ws,
                                                int n) {
  extern __shared__ float smem[];
  float* dvals = smem;           // n floats: d2_i
  float* red   = smem + n;       // 64 floats: 2 partials per wave
  float* bcast = smem + n + 64;  // 1 float: broadcast of new c / scale

  const int t    = threadIdx.x;
  const int lane = t & 31;
  const int wave = t >> 5;
  const int b    = blockIdx.x;
  const int k    = kptr[0];

  const float filled = ws[0];
  const float Q      = ws[1];
  const float twoQ   = 2.0f * Q;

  // ---- stage: d2_i = (2*s_eff - 1)*Q into LDS (coalesced float4) ----
  const float4* grow = reinterpret_cast<const float4*>(scores + (size_t)b * (size_t)n);
  float4* drow4 = reinterpret_cast<float4*>(dvals);
  const int nv4 = n >> 2;
  for (int i = t; i < nv4; i += blockDim.x) {
    float4 v = grow[i];
    v.x = (v.x == neg_inf_f()) ? filled : v.x;
    v.y = (v.y == neg_inf_f()) ? filled : v.y;
    v.z = (v.z == neg_inf_f()) ? filled : v.z;
    v.w = (v.w == neg_inf_f()) ? filled : v.w;
    float4 d;
    d.x = fmaf(v.x, twoQ, -Q);
    d.y = fmaf(v.y, twoQ, -Q);
    d.z = fmaf(v.z, twoQ, -Q);
    d.w = fmaf(v.w, twoQ, -Q);
    drow4[i] = d;
  }
  __syncthreads();

  // all-ones f16 B matrix: D[m][n] += sum_k A[m][k]
  v16h bones;
#pragma unroll
  for (int i = 0; i < 16; ++i) bones[i] = (_Float16)1.0f;

  const float fn   = (float)n;
  const float Klog = fast_log2((float)k) - fast_log2((float)(n - k));
  float c2 = 0.0f;      // c * log2(e)
  float scale = 0.0f;

  const float4* d4 = reinterpret_cast<const float4*>(dvals);
  const int waveBase4 = wave * (n >> 7);   // per-wave segment, in float4 units
  const int chunks    = (n >> 5) >> 9;     // 512-element chunks per wave (4 for n=65536)

  for (int it = 0; it < 200; ++it) {
    v8f acc = {};
    for (int m = 0; m < chunks; ++m) {
      v16h a;
#pragma unroll
      for (int s = 0; s < 4; ++s) {
        float4 dv = d4[waveBase4 + m * 128 + s * 32 + lane];  // ds_load_b128, conflict-free
        float s0 = fast_rcp(1.0f + fast_exp2(dv.x - c2));
        float s1 = fast_rcp(1.0f + fast_exp2(dv.y - c2));
        float s2 = fast_rcp(1.0f + fast_exp2(dv.z - c2));
        float s3 = fast_rcp(1.0f + fast_exp2(dv.w - c2));
        a[4 * s + 0] = (_Float16)s0;
        a[4 * s + 1] = (_Float16)s1;
        a[4 * s + 2] = (_Float16)s2;
        a[4 * s + 3] = (_Float16)s3;
      }
      // accumulate 512 sigmoid values into the f32 tile via the matrix pipe
      acc = __builtin_amdgcn_wmma_f32_16x16x32_f16(false, a, false, bones,
                                                   (short)0, acc, false, false);
    }
    // B=ones => every column of D identical; wave total = sum_m D[m][0]
    //  = (sum of 8 VGPRs on lane 0: rows 0..7) + (same on lane 16: rows 8..15)
    float p8 = acc[0] + acc[1] + acc[2] + acc[3] + acc[4] + acc[5] + acc[6] + acc[7];
    if (lane == 0)  red[2 * wave]     = p8;
    if (lane == 16) red[2 * wave + 1] = p8;
    __syncthreads();
    if (wave == 0) {
      float v = red[2 * lane] + red[2 * lane + 1];
      v += __shfl_xor(v, 16);
      v += __shfl_xor(v, 8);
      v += __shfl_xor(v, 4);
      v += __shfl_xor(v, 2);
      v += __shfl_xor(v, 1);
      if (lane == 0) {
        if (it < 199) bcast[0] = c2 + Klog + fast_log2(fn - v) - fast_log2(v);
        else          bcast[0] = (float)k * fast_rcp(v);  // final normalization k/S
      }
    }
    __syncthreads();
    if (it < 199) c2 = bcast[0];
    else          scale = bcast[0];
  }

  // ---- output: A_i = (k/S) * sigmoid(c - d_i) ----
  float4* orow = reinterpret_cast<float4*>(out + (size_t)b * (size_t)n);
  for (int i = t; i < nv4; i += blockDim.x) {
    float4 dv = d4[i];
    float4 o;
    o.x = scale * fast_rcp(1.0f + fast_exp2(dv.x - c2));
    o.y = scale * fast_rcp(1.0f + fast_exp2(dv.y - c2));
    o.z = scale * fast_rcp(1.0f + fast_exp2(dv.z - c2));
    o.w = scale * fast_rcp(1.0f + fast_exp2(dv.w - c2));
    orow[i] = o;
  }
}

extern "C" void kernel_launch(void* const* d_in, const int* in_sizes, int n_in,
                              void* d_out, int out_size, void* d_ws, size_t ws_size,
                              hipStream_t stream) {
  (void)n_in; (void)out_size; (void)ws_size;
  const float* scores = (const float*)d_in[0];
  const int*   kp     = (const int*)d_in[1];
  float* out = (float*)d_out;
  float* ws  = (float*)d_ws;

  const long long total = (long long)in_sizes[0];
  const int bs = 128;
  const int n  = (int)(total / bs);   // 65536

  sk_minmax<<<256, 256, 0, stream>>>(scores, total, ws);
  sk_scalars<<<1, 256, 0, stream>>>(ws);

  // row (n f32) + reduction scratch resident in LDS: 256 KB + 320 B <= 320 KB WGP LDS
  size_t sh = ((size_t)n + 80) * sizeof(float);
  sk_rows<<<bs, 1024, sh, stream>>>(scores, kp, out, ws, n);
}
